// FaSNet_origin_1443109012311
// MI455X (gfx1250) — compile-verified, matrix operand order checked by reading, not run on hardware
//
#include <hip/hip_runtime.h>
#include <hip/hip_bf16.h>

// ---------------------------------------------------------------------------
// FaSNet on MI455X (gfx1250): WMMA-based implementation.
//  * All dense contractions -> v_wmma_f32_16x16x32_f16 (fp32 accum).
//  * Recurrent LSTM cell: persistent kernel, Whh DMA'd into LDS by the
//    Tensor Data Mover (TENSOR_LOAD_TO_LDS, TENSORcnt), 16-batch tile per
//    workgroup, gates via WMMA from LDS.
//  * Correlations / norms / overlap-add: VALU (sub-1% of FLOPs).
// ---------------------------------------------------------------------------

typedef __attribute__((ext_vector_type(16))) _Float16 v16h;
typedef __attribute__((ext_vector_type(8)))  _Float16 v8h;
typedef __attribute__((ext_vector_type(8)))  float    v8f;
typedef __attribute__((ext_vector_type(4)))  unsigned int v4u;
typedef __attribute__((ext_vector_type(8)))  int      v8i;
typedef __attribute__((ext_vector_type(4)))  int      v4i;

// ---- problem constants (from the reference) --------------------------------
#define LLc     2002      // number of chunks L
#define FRAMEc  576
#define FILTc   513
#define INPc    608       // IN_DIM (577) padded to mult-32
#define SEGc    50
#define KSEGc   82        // segments after split_feature
#define BHc     2         // batch
#define NMc     4         // mics
#define TTc     64000
#define NSPc    2
#define CTXc    256
#define EPSc    1e-8f

__device__ __forceinline__ float sigm_(float x) { return 1.f / (1.f + __expf(-x)); }

// ---- WMMA helpers (layouts per CDNA5 ISA 7.12.2, wave32) -------------------
__device__ __forceinline__ v8f wmma_f16(v16h a, v16h b, v8f c) {
  return __builtin_amdgcn_wmma_f32_16x16x32_f16(false, a, false, b, (short)0, c,
                                                false, false);
}

// A (and B^T, i.e. weight rows) fragment: lane L<16 holds row M=L, K=kb..kb+7
// in halves 0..7 and K=kb+16..kb+23 in halves 8..15, kb = (L>=16)*8.
__device__ __forceinline__ v16h load_frag16(const _Float16* __restrict__ tile,
                                            int ld, int lane) {
  const int row = lane & 15;
  const int kb  = (lane >> 4) << 3;
  const _Float16* p = tile + (size_t)row * ld + kb;
  v8h lo = *(const v8h*)(p);
  v8h hi = *(const v8h*)(p + 16);
  v16h r;
#pragma unroll
  for (int j = 0; j < 8; ++j) { r[j] = lo[j]; r[j + 8] = hi[j]; }
  return r;
}

// ---- generic WMMA GEMM: C[M,N] = act(A[M,K] * W[N,K]^T + bias) -------------
// 128 threads = 4 waves; wave w computes a 16x16 tile at (blockIdx.x*16,
// (blockIdx.y*4+w)*16). M padded to 16, W padded to mult-64 rows, K mult-32.
__global__ void k_gemm(const _Float16* __restrict__ A, int lda,
                       const _Float16* __restrict__ W, int ldw,
                       const float* __restrict__ bias,
                       float* __restrict__ C, int ldc,
                       int N, int K, int act) {
  const int lane = threadIdx.x & 31;
  const int wave = threadIdx.x >> 5;
  const int m0 = blockIdx.x * 16;
  const int n0 = (blockIdx.y * 4 + wave) * 16;
  const _Float16* Ap = A + (size_t)m0 * lda;
  const _Float16* Wp = W + (size_t)n0 * ldw;
  v8f acc = {};
  for (int k0 = 0; k0 < K; k0 += 32) {
    v16h a = load_frag16(Ap + k0, lda, lane);
    v16h b = load_frag16(Wp + k0, ldw, lane);
    acc = wmma_f16(a, b, acc);
  }
  const int n = n0 + (lane & 15);
  if (n >= N) return;
  const float bv = bias ? bias[n] : 0.f;
  const int mr = (lane >> 4) << 3;
#pragma unroll
  for (int v = 0; v < 8; ++v) {
    float x = acc[v] + bv;
    if (act == 1) x = tanhf(x);
    else if (act == 2) x = sigm_(x);
    C[(size_t)(m0 + mr + v) * ldc + n] = x;
  }
}

// ---- persistent LSTM scan: gates via WMMA from LDS-resident Whh ------------
// block = 128 threads (4 waves) = one 16-row batch tile.
// LDS: Whh[512][128] f16 (131072B, staged by TDM) + gates[16][512] f32
// (32768B) + H[16][128] f16 (4096B) = 167936B dynamic (< 320KB WGP limit).
__global__ void k_lstm_scan(const _Float16* __restrict__ Whh,
                            const float* __restrict__ XG,
                            _Float16* __restrict__ HS, int ldh,
                            int batch, int T, int rev) {
  extern __shared__ char smem[];
  _Float16* sW = (_Float16*)smem;                              // [512][128]
  float*    sG = (float*)(smem + 512 * 128 * 2);               // [16][512]
  _Float16* sH = (_Float16*)(smem + 512 * 128 * 2 + 16 * 512 * 4); // [16][128]
  const int lane = threadIdx.x & 31;
  const int wave = threadIdx.x >> 5;
  const int b0 = blockIdx.x * 16;

#if defined(__gfx1250__)
  // Stage Whh (131072B contiguous) into LDS with the Tensor Data Mover.
  // D# group0: count=1, lds_addr=0 (dynamic-LDS base), global_addr, type=2.
  // D# group1: data_size=8B, tensor_dim0 = tile_dim0 = 16384 elems,
  //            tensor_dim0_stride = 16384, 1-D tile (tile_dim1/2 unused).
  if (wave == 0) {
    unsigned long long ga = (unsigned long long)(uintptr_t)Whh;
    v4u g0 = { 1u,                               // count=1, user mode
               0u,                               // lds_addr (dyn LDS base)
               (unsigned)(ga & 0xFFFFFFFFu),
               (unsigned)(((ga >> 32) & 0x01FFFFFFu) | (2u << 30)) };
    v8i g1 = { 0x00030000,                       // data_size=3 (8B), mask=0
               0x40000000,                       // tensor_dim0[15:0]=0x4000
               0x00010000,                       // tensor_dim1=1
               0x40000000,                       // tile_dim0=0x4000
               0,                                // tile_dim1/2 unused
               16384,                            // tensor_dim0_stride
               0, 0 };
    v4i gz = { 0, 0, 0, 0 };
#if __clang_major__ >= 23
    v8i gz8 = { 0, 0, 0, 0, 0, 0, 0, 0 };
    __builtin_amdgcn_tensor_load_to_lds(g0, g1, gz, gz, gz8, 0);
#else
    __builtin_amdgcn_tensor_load_to_lds(g0, g1, gz, gz, 0);
#endif
    __builtin_amdgcn_s_wait_tensorcnt(0);
  }
#else
  for (int i = threadIdx.x; i < 512 * 128 / 8; i += 128)
    ((v8h*)sW)[i] = ((const v8h*)Whh)[i];
#endif
  for (int i = threadIdx.x; i < 16 * 128; i += 128) sH[i] = (_Float16)0.f;
  __syncthreads();

  float c[16];
#pragma unroll
  for (int j = 0; j < 16; ++j) c[j] = 0.f;

  for (int tt = 0; tt < T; ++tt) {
    const int t = rev ? (T - 1 - tt) : tt;
    // gates = H @ Whh^T (+ precomputed XG): wave covers 128 gate columns.
#pragma unroll
    for (int nt = 0; nt < 8; ++nt) {
      const int n0 = wave * 128 + nt * 16;
      v8f acc = {};
#pragma unroll
      for (int kc = 0; kc < 4; ++kc) {
        v16h a = load_frag16(sH + kc * 32, 128, lane);
        v16h b = load_frag16(sW + (size_t)n0 * 128 + kc * 32, 128, lane);
        acc = wmma_f16(a, b, acc);
      }
      const int n = n0 + (lane & 15);
      const int mr = (lane >> 4) << 3;
#pragma unroll
      for (int v = 0; v < 8; ++v) {
        const int m = mr + v;
        float xg = (b0 + m < batch)
                       ? XG[((size_t)(b0 + m) * T + t) * 512 + n] : 0.f;
        sG[m * 512 + n] = acc[v] + xg;
      }
    }
    __syncthreads();
    // cell update: 2048 (m,h) cells / 128 threads = 16 each, c in registers.
#pragma unroll
    for (int j = 0; j < 16; ++j) {
      const int cell = threadIdx.x + j * 128;
      const int m = cell >> 7, h = cell & 127;
      const float gi = sG[m * 512 + h];
      const float gf = sG[m * 512 + 128 + h];
      const float gg = sG[m * 512 + 256 + h];
      const float go = sG[m * 512 + 384 + h];
      const float cn = sigm_(gf) * c[j] + sigm_(gi) * tanhf(gg);
      c[j] = cn;
      const float hh = sigm_(go) * tanhf(cn);
      sH[m * 128 + h] = (_Float16)hh;
      if (b0 + m < batch)
        HS[((size_t)(b0 + m) * T + t) * ldh + h] = (_Float16)hh;
    }
    __syncthreads();
  }
}

// ---- elementwise / gather / norm kernels -----------------------------------
__global__ void k_convert(const float* __restrict__ src, _Float16* __restrict__ dst,
                          int srows, int scols, int dcols, size_t total) {
  size_t i = (size_t)blockIdx.x * 256 + threadIdx.x;
  if (i >= total) return;
  int c = (int)(i % dcols);
  size_t r = i / dcols;
  float v = (r < (size_t)srows && c < scols) ? src[r * scols + c] : 0.f;
  dst[i] = (_Float16)v;
}

__global__ void k_bias_sum(const float* __restrict__ a, const float* __restrict__ b,
                           float* __restrict__ o, int n) {
  int i = blockIdx.x * 256 + threadIdx.x;
  if (i < n) o[i] = a[i] + b[i];
}

// build all_ctx [B, NM, L, FRAME] in f32 and f16 (zero-padded outside signal)
__global__ void k_chunk(const float* __restrict__ x, float* __restrict__ c32,
                        _Float16* __restrict__ c16, size_t total) {
  size_t i = (size_t)blockIdx.x * 256 + threadIdx.x;
  if (i >= total) return;
  int f = (int)(i % FRAMEc);
  size_t r = i / FRAMEc;
  int l = (int)(r % LLc); r /= LLc;
  int mic = (int)(r % NMc);
  int b = (int)(r / NMc);
  long pos = (long)l * 32 + f - (CTXc + 32);
  float v = (pos >= 0 && pos < TTc)
                ? x[((size_t)b * NMc + mic) * TTc + pos] : 0.f;
  c32[i] = v;
  c16[i] = (_Float16)v;
}

// ref cosine features: feat[b, l, 64+lag] = mean_mic cos(ref window, other seg)
__global__ void k_cossim_ref(const float* __restrict__ ctx, float* __restrict__ feat) {
  const int l = blockIdx.x % LLc;
  const int b = blockIdx.x / LLc;
  __shared__ float tn[3];
  const float* refp = ctx + (((size_t)b * NMc) * LLc + l) * FRAMEc;
  if (threadIdx.x < 3) {
    const float* tg = ctx + (((size_t)b * NMc + 1 + threadIdx.x) * LLc + l) * FRAMEc + CTXc;
    float s = 0;
    for (int j = 0; j < 64; ++j) { float v = tg[j]; s += v * v; }
    tn[threadIdx.x] = sqrtf(s) + EPSc;
  }
  __syncthreads();
  const int lag = threadIdx.x;
  float* orow = feat + ((size_t)b * LLc + l) * INPc;
  if (lag >= FILTc) { if (64 + lag < INPc) orow[64 + lag] = 0.f; return; }
  float re = 0;
  for (int j = 0; j < 64; ++j) { float v = refp[lag + j]; re += v * v; }
  const float rn = sqrtf(re) + EPSc;
  float acc = 0;
  for (int mm = 0; mm < 3; ++mm) {
    const float* tg = ctx + (((size_t)b * NMc + 1 + mm) * LLc + l) * FRAMEc + CTXc;
    float d = 0;
    for (int j = 0; j < 64; ++j) d += refp[lag + j] * tg[j];
    acc += d / (rn * tn[mm]);
  }
  orow[64 + lag] = acc * (1.f / 3.f);
}

// other-path cosine features vs beamformed ref output
__global__ void k_cossim_other(const float* __restrict__ ctx,
                               const float* __restrict__ refout,
                               float* __restrict__ feat) {
  const int l = blockIdx.x % LLc;
  const int bS = blockIdx.x / LLc;   // b*NSP+spk
  const int b = bS >> 1;
  __shared__ float tn;
  const float* tgt = refout + ((size_t)bS * LLc + l) * 64;
  if (threadIdx.x == 0) {
    float s = 0;
    for (int j = 0; j < 64; ++j) s += tgt[j] * tgt[j];
    tn = sqrtf(s) + EPSc;
  }
  __syncthreads();
  const int lag = threadIdx.x;
  for (int mm = 0; mm < 3; ++mm) {
    float* orow = feat + (((size_t)bS * 3 + mm) * LLc + l) * INPc;
    if (lag >= FILTc) { if (64 + lag < INPc) orow[64 + lag] = 0.f; continue; }
    const float* rp = ctx + (((size_t)b * NMc + 1 + mm) * LLc + l) * FRAMEc;
    float re = 0, d = 0;
    for (int j = 0; j < 64; ++j) {
      float v = rp[lag + j];
      re += v * v;
      d += v * tgt[j];
    }
    orow[64 + lag] = d / ((sqrtf(re) + EPSc) * tn);
  }
}

// group-norm stats (mean, rstd) per batch slice
__global__ void k_gn_stats(const float* __restrict__ x, float* __restrict__ stats,
                           int rowsPerB, int rowStride, int cols) {
  const int b = blockIdx.x;
  const float* base = x + (size_t)b * rowsPerB * rowStride;
  const int total = rowsPerB * cols;
  float s = 0, ss = 0;
  for (int i = threadIdx.x; i < total; i += blockDim.x) {
    int r = i / cols, c = i % cols;
    float v = base[(size_t)r * rowStride + c];
    s += v; ss += v * v;
  }
  __shared__ float sh[512];
  sh[threadIdx.x] = s; sh[256 + threadIdx.x] = ss;
  __syncthreads();
  for (int o = 128; o > 0; o >>= 1) {
    if ((int)threadIdx.x < o) {
      sh[threadIdx.x] += sh[threadIdx.x + o];
      sh[256 + threadIdx.x] += sh[256 + threadIdx.x + o];
    }
    __syncthreads();
  }
  if (threadIdx.x == 0) {
    float mu = sh[0] / total;
    float var = sh[256] / total - mu * mu;
    stats[2 * b] = mu;
    stats[2 * b + 1] = rsqrtf((var > 0.f ? var : 0.f) + EPSc);
  }
}

__global__ void k_gn_apply(float* __restrict__ x, const float* __restrict__ stats,
                           const float* __restrict__ g, const float* __restrict__ bt,
                           int rowsPerB, int rowStride, int cols, size_t total) {
  size_t i = (size_t)blockIdx.x * 256 + threadIdx.x;
  if (i >= total) return;
  int c = (int)(i % cols);
  size_t r = i / cols;
  int row = (int)(r % rowsPerB);
  int b = (int)(r / rowsPerB);
  size_t off = ((size_t)b * rowsPerB + row) * rowStride + c;
  x[off] = (x[off] - stats[2 * b]) * stats[2 * b + 1] * g[c] + bt[c];
}

// residual add of group-normed R into S; mode 1: R is [b,s,k,n], S is [b,k,s,n]
__global__ void k_gn_add(const float* __restrict__ R, const float* __restrict__ stats,
                         const float* __restrict__ g, const float* __restrict__ bt,
                         float* __restrict__ S, int mode, size_t total) {
  size_t i = (size_t)blockIdx.x * 256 + threadIdx.x;
  if (i >= total) return;
  int n = (int)(i & 63);
  size_t r = i >> 6;
  int s = (int)(r % SEGc); r /= SEGc;
  int k = (int)(r % KSEGc);
  int b = (int)(r / KSEGc);
  size_t src = mode ? ((((size_t)b * SEGc + s) * KSEGc + k) * 64 + n) : i;
  S[i] += (R[src] - stats[2 * b]) * stats[2 * b + 1] * g[n] + bt[n];
}

// split_feature: S[b,k,s,n] from bottleneck out0[(b*L+t), n]
__global__ void k_split(const float* __restrict__ out0, float* __restrict__ S,
                        size_t total) {
  size_t i = (size_t)blockIdx.x * 256 + threadIdx.x;
  if (i >= total) return;
  int n = (int)(i & 63);
  size_t r = i >> 6;
  int s = (int)(r % SEGc); r /= SEGc;
  int k = (int)(r % KSEGc);
  int b = (int)(r / KSEGc);
  int t = (k & 1) ? ((k >> 1) * SEGc + s) : ((k >> 1) * SEGc + s - 25);
  S[i] = (t >= 0 && t < LLc) ? out0[((size_t)b * LLc + t) * 64 + n] : 0.f;
}

// col-major LSTM input gather: X[(b*50+s), k, n] = S[b,k,s,n], f16, zero pad
__global__ void k_colgather(const float* __restrict__ S, _Float16* __restrict__ X,
                            int Mreal, size_t total) {
  size_t i = (size_t)blockIdx.x * 256 + threadIdx.x;
  if (i >= total) return;
  int n = (int)(i & 63);
  size_t row = i >> 6;
  if (row >= (size_t)Mreal) { X[i] = (_Float16)0.f; return; }
  int k = (int)(row % KSEGc); row /= KSEGc;
  int s = (int)(row % SEGc);
  int b = (int)(row / SEGc);
  X[i] = (_Float16)S[(((size_t)b * KSEGc + k) * SEGc + s) * 64 + n];
}

__global__ void k_prelu(float* __restrict__ S, const float* __restrict__ a,
                        size_t total) {
  size_t i = (size_t)blockIdx.x * 256 + threadIdx.x;
  if (i >= total) return;
  float x = S[i];
  S[i] = x >= 0.f ? x : a[0] * x;
}

// merge_feature -> Fm16[(b*nspk+spk)*L + j, c] (f16), pad rows zeroed
__global__ void k_merge(const float* __restrict__ O, _Float16* __restrict__ Fm,
                        int nspk, int Mreal, size_t total) {
  size_t i = (size_t)blockIdx.x * 256 + threadIdx.x;
  if (i >= total) return;
  int c = (int)(i & 63);
  size_t row = i >> 6;
  if (row >= (size_t)Mreal) { Fm[i] = (_Float16)0.f; return; }
  int j = (int)(row % LLc);
  int bs = (int)(row / LLc);
  int spk = bs % nspk;
  int b = bs / nspk;
  int ke = 2 * ((j + 25) / SEGc), se = (j + 25) % SEGc;
  int ko = 2 * (j / SEGc) + 1,    so = j % SEGc;
  float v = O[(((size_t)b * KSEGc + ke) * SEGc + se) * 128 + spk * 64 + c] +
            O[(((size_t)b * KSEGc + ko) * SEGc + so) * 128 + spk * 64 + c];
  Fm[i] = (_Float16)v;
}

__global__ void k_mul(const float* __restrict__ a, const float* __restrict__ b,
                      float* __restrict__ o, size_t total) {
  size_t i = (size_t)blockIdx.x * 256 + threadIdx.x;
  if (i >= total) return;
  o[i] = a[i] * b[i];
}

// apply learned filter: out[bS,l,w] = sum_j ctx[b,mic0,l,w+j]*filt[bS,l,j]
__global__ void k_applyfilt_ref(const float* __restrict__ ctx,
                                const float* __restrict__ filt,
                                float* __restrict__ outw, size_t total) {
  size_t i = (size_t)blockIdx.x * 256 + threadIdx.x;
  if (i >= total) return;
  int w = (int)(i & 63);
  size_t r = i >> 6;
  int l = (int)(r % LLc);
  int bS = (int)(r / LLc);
  int b = bS >> 1;
  const float* rp = ctx + (((size_t)b * NMc) * LLc + l) * FRAMEc + w;
  const float* fp = filt + ((size_t)bS * LLc + l) * 528;
  __builtin_prefetch(fp, 0, 1);
  __builtin_prefetch(rp, 0, 1);
  float s = 0;
  for (int j = 0; j < FILTc; ++j) s += rp[j] * fp[j];
  outw[i] = s;
}

__global__ void k_applyfilt_other(const float* __restrict__ ctx,
                                  const float* __restrict__ filt,
                                  float* __restrict__ outw, size_t total) {
  size_t i = (size_t)blockIdx.x * 256 + threadIdx.x;
  if (i >= total) return;
  int w = (int)(i & 63);
  size_t r = i >> 6;
  int l = (int)(r % LLc); r /= LLc;
  int mm = (int)(r % 3);
  int bS = (int)(r / 3);
  int b = bS >> 1;
  const float* rp = ctx + (((size_t)b * NMc + 1 + mm) * LLc + l) * FRAMEc + w;
  const float* fp = filt + (((size_t)bS * 3 + mm) * LLc + l) * 528;
  float s = 0;
  for (int j = 0; j < FILTc; ++j) s += rp[j] * fp[j];
  outw[i] = s;
}

// final overlap-add + mean over mics -> out[b, spk, t]
__global__ void k_final(const float* __restrict__ refout,
                        const float* __restrict__ otherout,
                        float* __restrict__ out, size_t total) {
  size_t i = (size_t)blockIdx.x * 256 + threadIdx.x;
  if (i >= total) return;
  int t = (int)(i % TTc);
  int bS = (int)(i / TTc);
  int le = 2 * ((t + 32) / 64), pe = (t + 32) % 64;
  int lo = 2 * (t / 64) + 1,    po = t % 64;
  float acc = refout[((size_t)bS * LLc + le) * 64 + pe] +
              refout[((size_t)bS * LLc + lo) * 64 + po];
  for (int mm = 0; mm < 3; ++mm) {
    const float* ob = otherout + ((size_t)bS * 3 + mm) * LLc * 64;
    acc += ob[(size_t)le * 64 + pe] + ob[(size_t)lo * 64 + po];
  }
  out[i] = acc * 0.25f;
}

// ---------------------------------------------------------------------------
// Host-side orchestration
// ---------------------------------------------------------------------------
static inline int ceil16i(int x) { return (x + 15) & ~15; }
static inline dim3 g1(size_t total) { return dim3((unsigned)((total + 255) / 256)); }

struct LstmP {
  const float *Whh_b, *Whh_f, *Wih_b, *Wih_f, *bhh_b, *bhh_f, *bih_b, *bih_f;
  const float *norm_b, *norm_g, *proj_W, *proj_b;
};
struct BFP {
  const float *BN_W, *gate_W, *gate_b;
  LstmP col[4], row[4];
  const float *out_W, *out_b, *outp_W, *outp_b, *prelu_a;
};
struct Scratch {
  float *out0, *S, *XG, *R, *O, *FT, *GT, *filt, *bias512, *stats;
  _Float16 *X16, *Hcat, *Fm16, *w16a, *w16b;
};

static LstmP readL(void* const* din, int& ix) {
  LstmP p;
  p.Whh_b = (const float*)din[ix++]; p.Whh_f = (const float*)din[ix++];
  p.Wih_b = (const float*)din[ix++]; p.Wih_f = (const float*)din[ix++];
  p.bhh_b = (const float*)din[ix++]; p.bhh_f = (const float*)din[ix++];
  p.bih_b = (const float*)din[ix++]; p.bih_f = (const float*)din[ix++];
  p.norm_b = (const float*)din[ix++]; p.norm_g = (const float*)din[ix++];
  p.proj_W = (const float*)din[ix++]; p.proj_b = (const float*)din[ix++];
  return p;
}
static BFP readBF(void* const* din, int& ix) {
  BFP q;
  q.BN_W = (const float*)din[ix++];
  q.gate_W = (const float*)din[ix++];
  q.gate_b = (const float*)din[ix++];
  for (int l = 0; l < 4; ++l) { q.col[l] = readL(din, ix); q.row[l] = readL(din, ix); }
  q.out_W = (const float*)din[ix++];
  q.out_b = (const float*)din[ix++];
  q.outp_W = (const float*)din[ix++];
  q.outp_b = (const float*)din[ix++];
  q.prelu_a = (const float*)din[ix++];
  return q;
}

static void conv(hipStream_t st, const float* src, _Float16* dst,
                 int sr, int sc, int dr, int dc) {
  size_t total = (size_t)dr * dc;
  k_convert<<<g1(total), 256, 0, st>>>(src, dst, sr, sc, dc, total);
}
static void gemm(hipStream_t st, const _Float16* A, int lda, const _Float16* W,
                 int ldw, const float* bias, float* C, int ldc,
                 int Mp, int N, int NP64, int K, int act) {
  dim3 g(Mp / 16, NP64 / 64);
  k_gemm<<<g, 128, 0, st>>>(A, lda, W, ldw, bias, C, ldc, N, K, act);
}
static void scan(hipStream_t st, const _Float16* Whh16, const float* XG,
                 _Float16* Hbase, int ldh, int batch, int T, int rev) {
  int blocks = (batch + 15) / 16;
  size_t sm = 512 * 128 * 2 + 16 * 512 * 4 + 16 * 128 * 2;  // 167936B
  k_lstm_scan<<<blocks, 128, sm, st>>>(Whh16, XG, Hbase, ldh, batch, T, rev);
}

static void bilstm(hipStream_t st, Scratch& w, const _Float16* X16, int batch,
                   int T, const LstmP& P, float* R, int Mxp) {
  // forward
  k_bias_sum<<<g1(512), 256, 0, st>>>(P.bih_f, P.bhh_f, w.bias512, 512);
  conv(st, P.Wih_f, w.w16b, 512, 64, 512, 64);
  gemm(st, X16, 64, w.w16b, 64, w.bias512, w.XG, 512, Mxp, 512, 512, 64, 0);
  conv(st, P.Whh_f, w.w16a, 512, 128, 512, 128);
  scan(st, w.w16a, w.XG, w.Hcat, 256, batch, T, 0);
  // backward
  k_bias_sum<<<g1(512), 256, 0, st>>>(P.bih_b, P.bhh_b, w.bias512, 512);
  conv(st, P.Wih_b, w.w16b, 512, 64, 512, 64);
  gemm(st, X16, 64, w.w16b, 64, w.bias512, w.XG, 512, Mxp, 512, 512, 64, 0);
  conv(st, P.Whh_b, w.w16a, 512, 128, 512, 128);
  scan(st, w.w16a, w.XG, w.Hcat + 128, 256, batch, T, 1);
  // projection 256 -> 64
  conv(st, P.proj_W, w.w16b, 64, 256, 64, 256);
  gemm(st, w.Hcat, 256, w.w16b, 256, P.proj_b, R, 64, Mxp, 64, 64, 256, 0);
}

static void run_bf(hipStream_t st, const BFP& P, const _Float16* feat16, int Bb,
                   int nspk, Scratch& w) {
  const int Mb = Bb * LLc, Mbp = ceil16i(Mb);
  // bottleneck
  conv(st, P.BN_W, w.w16b, 64, 577, 64, INPc);
  gemm(st, feat16, INPc, w.w16b, INPc, nullptr, w.out0, 64, Mbp, 64, 64, INPc, 0);
  const size_t Stot = (size_t)Bb * KSEGc * SEGc * 64;
  k_split<<<g1(Stot), 256, 0, st>>>(w.out0, w.S, Stot);
  const int batR = Bb * KSEGc, batC = Bb * SEGc;
  const int Mx = Bb * KSEGc * SEGc, Mxp = ceil16i(Mx);
  const size_t Xtot = (size_t)Mxp * 64;
  for (int lyr = 0; lyr < 4; ++lyr) {
    // row pass ([b,k] batch, T=SEG)
    conv(st, w.S, w.X16, Mx, 64, Mxp, 64);
    bilstm(st, w, w.X16, batR, SEGc, P.row[lyr], w.R, Mxp);
    k_gn_stats<<<Bb, 256, 0, st>>>(w.R, w.stats, KSEGc * SEGc, 64, 64);
    k_gn_add<<<g1(Stot), 256, 0, st>>>(w.R, w.stats, P.row[lyr].norm_g,
                                       P.row[lyr].norm_b, w.S, 0, Stot);
    // col pass ([b,s] batch, T=KSEG)
    k_colgather<<<g1(Xtot), 256, 0, st>>>(w.S, w.X16, Mx, Xtot);
    bilstm(st, w, w.X16, batC, KSEGc, P.col[lyr], w.R, Mxp);
    k_gn_stats<<<Bb, 256, 0, st>>>(w.R, w.stats, SEGc * KSEGc, 64, 64);
    k_gn_add<<<g1(Stot), 256, 0, st>>>(w.R, w.stats, P.col[lyr].norm_g,
                                       P.col[lyr].norm_b, w.S, 1, Stot);
  }
  k_prelu<<<g1(Stot), 256, 0, st>>>(w.S, P.prelu_a, Stot);
  // speaker expansion 64 -> 64*nspk
  conv(st, w.S, w.X16, Mx, 64, Mxp, 64);
  const int No = 64 * nspk, NoP = (No + 63) & ~63;
  conv(st, P.out_W, w.w16b, No, 64, NoP, 64);
  gemm(st, w.X16, 64, w.w16b, 64, P.out_b, w.O, 128, Mxp, No, NoP, 64, 0);
  // merge overlapped segments back to the chunk axis
  const int Mf = Bb * nspk * LLc, Mfp = ceil16i(Mf);
  const size_t Ftot = (size_t)Mfp * 64;
  k_merge<<<g1(Ftot), 256, 0, st>>>(w.O, w.Fm16, nspk, Mf, Ftot);
  // filter + gate heads (N=513 padded to 576 weight rows, ldc 528)
  conv(st, P.outp_W, w.w16b, FILTc, 64, 576, 64);
  gemm(st, w.Fm16, 64, w.w16b, 64, P.outp_b, w.FT, 528, Mfp, FILTc, 576, 64, 1);
  conv(st, P.gate_W, w.w16b, FILTc, 64, 576, 64);
  gemm(st, w.Fm16, 64, w.w16b, 64, P.gate_b, w.GT, 528, Mfp, FILTc, 576, 64, 2);
  const size_t Gtot = (size_t)Mfp * 528;
  k_mul<<<g1(Gtot), 256, 0, st>>>(w.FT, w.GT, w.filt, Gtot);
}

extern "C" void kernel_launch(void* const* d_in, const int* in_sizes, int n_in,
                              void* d_out, int out_size, void* d_ws, size_t ws_size,
                              hipStream_t stream) {
  (void)in_sizes; (void)n_in; (void)out_size; (void)ws_size;
  // inputs: x, num_mic, then params flattened in JAX pytree (sorted-key) order.
  const float* x = (const float*)d_in[0];
  int ix = 2;
  const float* enc_W = (const float*)d_in[ix++];
  const float* enc_ln_b = (const float*)d_in[ix++];
  const float* enc_ln_g = (const float*)d_in[ix++];
  BFP otherP = readBF(d_in, ix);
  BFP refP = readBF(d_in, ix);
  float* out = (float*)d_out;

  // ---- workspace carve-out ----
  struct Bump { char* p; size_t off; } bp{(char*)d_ws, 0};
  auto alloc = [&](size_t bytes) -> void* {
    size_t a = (bp.off + 255) & ~(size_t)255;
    bp.off = a + bytes;
    return bp.p + a;
  };
  const size_t ctxRows = (size_t)BHc * NMc * LLc + 16;
  float* ctx32 = (float*)alloc(ctxRows * FRAMEc * 4);
  _Float16* ctx16 = (_Float16*)alloc(ctxRows * FRAMEc * 2);
  float* featR = (float*)alloc(((size_t)2 * LLc + 16) * INPc * 4);
  _Float16* feat16R = (_Float16*)alloc((size_t)ceil16i(2 * LLc) * INPc * 2);
  float* featO = (float*)alloc(((size_t)12 * LLc + 16) * INPc * 4);
  _Float16* feat16O = (_Float16*)alloc((size_t)ceil16i(12 * LLc) * INPc * 2);
  float* refout = (float*)alloc((size_t)4 * LLc * 64 * 4);
  float* otherout = (float*)alloc((size_t)12 * LLc * 64 * 4);
  const int MxMax = ceil16i(12 * KSEGc * SEGc);   // 49200
  const int MfMax = ceil16i(12 * LLc);
  Scratch w;
  w.out0 = (float*)alloc((size_t)MfMax * 64 * 4);
  w.S = (float*)alloc((size_t)12 * KSEGc * SEGc * 64 * 4);
  w.XG = (float*)alloc((size_t)MxMax * 512 * 4);
  w.R = (float*)alloc((size_t)MxMax * 64 * 4);
  w.O = (float*)alloc((size_t)MxMax * 128 * 4);
  w.FT = (float*)alloc((size_t)MfMax * 528 * 4);
  w.GT = (float*)alloc((size_t)MfMax * 528 * 4);
  w.filt = (float*)alloc((size_t)MfMax * 528 * 4);
  w.bias512 = (float*)alloc(512 * 4);
  w.stats = (float*)alloc(2 * 12 * 4);
  w.X16 = (_Float16*)alloc((size_t)MxMax * 64 * 2);
  w.Hcat = (_Float16*)alloc((size_t)MxMax * 256 * 2);
  w.Fm16 = (_Float16*)alloc((size_t)MfMax * 64 * 2);
  w.w16a = (_Float16*)alloc((size_t)512 * 128 * 2);
  w.w16b = (_Float16*)alloc((size_t)64 * INPc * 2 + 576 * 64 * 2);

  // 1) chunk all mics into context frames (f32 + f16 copies)
  const size_t ctot = (size_t)BHc * NMc * LLc * FRAMEc;
  k_chunk<<<g1(ctot), 256, 0, stream>>>(x, ctx32, ctx16, ctot);

  // 2) reference-mic encoder features + group norm
  conv(stream, enc_W, w.w16b, 64, FRAMEc, 64, FRAMEc);
  for (int b = 0; b < BHc; ++b) {
    const _Float16* A = ctx16 + ((size_t)b * NMc) * LLc * FRAMEc;
    gemm(stream, A, FRAMEc, w.w16b, FRAMEc, nullptr,
         featR + (size_t)b * LLc * INPc, INPc, ceil16i(LLc), 64, 64, FRAMEc, 0);
  }
  k_gn_stats<<<BHc, 256, 0, stream>>>(featR, w.stats, LLc, INPc, 64);
  k_gn_apply<<<g1((size_t)BHc * LLc * 64), 256, 0, stream>>>(
      featR, w.stats, enc_ln_g, enc_ln_b, LLc, INPc, 64, (size_t)BHc * LLc * 64);
  // 3) ref cosine features
  k_cossim_ref<<<BHc * LLc, 544, 0, stream>>>(ctx32, featR);
  conv(stream, featR, feat16R, 2 * LLc, INPc, ceil16i(2 * LLc), INPc);

  // 4) reference BF module -> filters, apply -> ref beamformed windows
  run_bf(stream, refP, feat16R, BHc, NSPc, w);
  k_applyfilt_ref<<<g1((size_t)4 * LLc * 64), 256, 0, stream>>>(
      ctx32, w.filt, refout, (size_t)4 * LLc * 64);

  // 5) other-mic path: encoder feats (12 slices), gn, cos vs refout
  conv(stream, enc_W, w.w16b, 64, FRAMEc, 64, FRAMEc);
  for (int b = 0; b < BHc; ++b)
    for (int spk = 0; spk < NSPc; ++spk)
      for (int mm = 0; mm < 3; ++mm) {
        int bi = (b * NSPc + spk) * 3 + mm;
        const _Float16* A = ctx16 + ((size_t)b * NMc + 1 + mm) * LLc * FRAMEc;
        gemm(stream, A, FRAMEc, w.w16b, FRAMEc, nullptr,
             featO + (size_t)bi * LLc * INPc, INPc, ceil16i(LLc), 64, 64,
             FRAMEc, 0);
      }
  k_gn_stats<<<12, 256, 0, stream>>>(featO, w.stats, LLc, INPc, 64);
  k_gn_apply<<<g1((size_t)12 * LLc * 64), 256, 0, stream>>>(
      featO, w.stats, enc_ln_g, enc_ln_b, LLc, INPc, 64, (size_t)12 * LLc * 64);
  k_cossim_other<<<4 * LLc, 544, 0, stream>>>(ctx32, refout, featO);
  conv(stream, featO, feat16O, 12 * LLc, INPc, ceil16i(12 * LLc), INPc);

  // 6) other BF module -> filters, apply per mic
  run_bf(stream, otherP, feat16O, 12, 1, w);
  k_applyfilt_other<<<g1((size_t)12 * LLc * 64), 256, 0, stream>>>(
      ctx32, w.filt, otherout, (size_t)12 * LLc * 64);

  // 7) overlap-add + mic mean
  k_final<<<g1((size_t)BHc * NSPc * TTc), 256, 0, stream>>>(
      refout, otherout, out, (size_t)BHc * NSPc * TTc);
}
